// SoftAlignment_27659589386627
// MI455X (gfx1250) — matrix operand order repaired and checked by reading, not compiled, not run on hardware
//
#include <hip/hip_runtime.h>
#include <hip/hip_bf16.h>

typedef _Float16 h4_t  __attribute__((ext_vector_type(4)));
typedef _Float16 h8_t  __attribute__((ext_vector_type(8)));
typedef _Float16 v16h  __attribute__((ext_vector_type(16)));
typedef float    v8f   __attribute__((ext_vector_type(8)));

#define B_DIM 64
#define L_DIM 512
#define H_DIM 1024
#define EPSV  1e-13f

// ---------------------------------------------------------------------------
// Pass 1: fused softmax + mask-renormalize, f32 -> f16 weights into workspace.
// One wave32 per row of 512. 8 waves (8 rows) per 256-thread block.
// ---------------------------------------------------------------------------
__global__ __launch_bounds__(256)
void softalign_softmax_kernel(const float* __restrict__ attn,
                              const float* __restrict__ mask,
                              _Float16* __restrict__ W) {
    const int wave = threadIdx.x >> 5;
    const int lane = threadIdx.x & 31;
    const long long row = (long long)blockIdx.x * 8 + wave;   // 0 .. B*L-1

    const float* __restrict__ a = attn + row * L_DIM;

    // Load 16 floats per lane as 4x float4, coalesced across the wave.
    float4 v[4];
    float mx = -__builtin_inff();
#pragma unroll
    for (int j = 0; j < 4; ++j) {
        v[j] = *(const float4*)(a + lane * 4 + j * 128);
        mx = fmaxf(mx, fmaxf(fmaxf(v[j].x, v[j].y), fmaxf(v[j].z, v[j].w)));
    }
    // wave32 max reduction
#pragma unroll
    for (int off = 16; off >= 1; off >>= 1)
        mx = fmaxf(mx, __shfl_xor(mx, off, 32));

    float e[4][4];
    float sum = 0.0f;
#pragma unroll
    for (int j = 0; j < 4; ++j) {
        e[j][0] = __expf(v[j].x - mx);
        e[j][1] = __expf(v[j].y - mx);
        e[j][2] = __expf(v[j].z - mx);
        e[j][3] = __expf(v[j].w - mx);
        sum += e[j][0] + e[j][1] + e[j][2] + e[j][3];
    }
#pragma unroll
    for (int off = 16; off >= 1; off >>= 1)
        sum += __shfl_xor(sum, off, 32);

    // softmax row sums to 1 -> masked row sums to m -> renorm scale = m/(m+EPS)
    const float m = mask[row];
    const float scale = (m / (m + EPSV)) / sum;

    _Float16* __restrict__ wout = W + row * L_DIM;
#pragma unroll
    for (int j = 0; j < 4; ++j) {
        h4_t h;
        h[0] = (_Float16)(e[j][0] * scale);
        h[1] = (_Float16)(e[j][1] * scale);
        h[2] = (_Float16)(e[j][2] * scale);
        h[3] = (_Float16)(e[j][3] * scale);
        *(h4_t*)(wout + lane * 4 + j * 128) = h;
    }
}

// ---------------------------------------------------------------------------
// Pass 2: batched GEMM  O[b] = W[b](512x512 f16) * S[b](512x1024 f32->f16)
// Block tile: M=128, N=64, K-step=32. 8 waves; wave w computes rows
// [m0+16w, m0+16w+16) x 64 cols as 4 WMMA 16x16 accumulators.
//
// A fragments: direct global b128 loads (row-major f16 W matches the 16-bit
//   A 16x32 layout: lane half selects K runs of 8).
// B fragments: each thread loads a column strip of 8 K-values (wave-coalesced
//   dword loads), packs to f16, and writes ONE ds_store_b128 directly in WMMA
//   B-fragment layout. Double-buffered LDS: global loads of tile k+1 issue
//   before the WMMA chain on tile k; one barrier per k-step.
// ---------------------------------------------------------------------------
#define TM 128
#define TN 64
#define TK 32
#define FPAD 24   // halfwords per fragment-lane row (16 used + 8 pad, 48B)

__global__ __launch_bounds__(256)
void softalign_gemm_kernel(const _Float16* __restrict__ W,
                           const float* __restrict__ S,
                           float* __restrict__ out) {
    __shared__ __align__(16) _Float16 FragB[2][4][32][FPAD];   // 12 KB

    const int b    = blockIdx.z;
    const int m0   = blockIdx.y * TM;
    const int n0   = blockIdx.x * TN;
    const int tid  = threadIdx.x;
    const int wave = tid >> 5;
    const int lane = tid & 31;
    const int hsel = lane >> 4;    // lane half: 0 or 1
    const int lidx = lane & 15;

    const _Float16* __restrict__ Wb = W   + (size_t)b * L_DIM * L_DIM;
    const float*    __restrict__ Sb = S   + (size_t)b * L_DIM * H_DIM;
    float*          __restrict__ Ob = out + (size_t)b * L_DIM * H_DIM;

    v8f acc[4] = {};

    // ---- staging-role indices (fixed per thread) ----
    const int nloc = tid & 63;                      // column in tile: 0..63
    const int kg   = tid >> 6;                      // K-group of 8: 0..3
    const int s_nt = nloc >> 4;                     // target subtile
    const int s_l  = (kg & 1) * 16 + (nloc & 15);   // target fragment lane
    const int s_j0 = (kg >> 1) * 8;                 // target element offset
    const float* s_src = Sb + (size_t)(kg * 8) * H_DIM + (n0 + nloc);

    const int arow = m0 + wave * 16 + lidx;         // A-fragment row
    const _Float16* ap = Wb + (size_t)arow * L_DIM + 8 * hsel;

    // ---- prologue: stage tile k=0 into buffer 0 ----
    {
        h8_t hv;
#pragma unroll
        for (int i = 0; i < 8; ++i)
            hv[i] = (_Float16)s_src[(size_t)i * H_DIM];
        *(h8_t*)&FragB[0][s_nt][s_l][s_j0] = hv;
        s_src += (size_t)TK * H_DIM;   // now points at tile k=TK
    }
    __syncthreads();

    for (int k0 = 0; k0 < L_DIM; k0 += TK) {
        const int  cur     = (k0 >> 5) & 1;
        const bool hasNext = (k0 + TK) < L_DIM;

        // ---- issue next-tile global loads first (overlap with WMMA) ----
        float f[8];
        if (hasNext) {
#pragma unroll
            for (int i = 0; i < 8; ++i)
                f[i] = s_src[(size_t)i * H_DIM];
            __builtin_prefetch(s_src + (size_t)TK * H_DIM, 0, 3);  // tile k+2
            __builtin_prefetch(ap + 2 * TK, 0, 3);
        }

        // ---- A fragment: 16-bit A 16x32 layout ----
        const h8_t a_lo = *(const h8_t*)(ap);
        const h8_t a_hi = *(const h8_t*)(ap + 16);
        ap += TK;
        v16h afrag;
#pragma unroll
        for (int j = 0; j < 8; ++j) { afrag[j] = a_lo[j]; afrag[j + 8] = a_hi[j]; }

        // ---- B fragments (2x ds_load_b128 each) + WMMA over 4 N-subtiles ----
#pragma unroll
        for (int nt = 0; nt < 4; ++nt) {
            const _Float16* bp = &FragB[cur][nt][lane][0];
            const h8_t b_lo = *(const h8_t*)(bp);
            const h8_t b_hi = *(const h8_t*)(bp + 8);
            v16h bfrag;
#pragma unroll
            for (int j = 0; j < 8; ++j) { bfrag[j] = b_lo[j]; bfrag[j + 8] = b_hi[j]; }

            acc[nt] = __builtin_amdgcn_wmma_f32_16x16x32_f16(
                /*neg_a=*/false, afrag, /*neg_b=*/false, bfrag,
                /*c_mod=*/(short)0, acc[nt], /*reuse_a=*/false, /*reuse_b=*/false);
        }

        // ---- convert + store next tile into the other buffer ----
        if (hasNext) {
            h8_t hv;
#pragma unroll
            for (int i = 0; i < 8; ++i)
                hv[i] = (_Float16)f[i];
            *(h8_t*)&FragB[cur ^ 1][s_nt][s_l][s_j0] = hv;
            s_src += (size_t)TK * H_DIM;
        }
        __syncthreads();
    }

    // ---- writeback: C/D f32 16x16 layout: elem r -> row r + 8*hsel, col lidx
#pragma unroll
    for (int nt = 0; nt < 4; ++nt) {
        const int col = n0 + nt * 16 + lidx;
#pragma unroll
        for (int r = 0; r < 8; ++r) {
            const int rowo = m0 + wave * 16 + r + 8 * hsel;
            Ob[(size_t)rowo * H_DIM + col] = acc[nt][r];
        }
    }
}

// ---------------------------------------------------------------------------
extern "C" void kernel_launch(void* const* d_in, const int* in_sizes, int n_in,
                              void* d_out, int out_size, void* d_ws, size_t ws_size,
                              hipStream_t stream) {
    (void)in_sizes; (void)n_in; (void)out_size; (void)ws_size;
    const float* sentence = (const float*)d_in[0];  // [64, 512, 1024] f32
    const float* mask     = (const float*)d_in[1];  // [64, 512]       f32
    const float* attn     = (const float*)d_in[2];  // [64, 512, 512]  f32
    float*       out      = (float*)d_out;          // [64, 512, 1024] f32
    _Float16*    Wf16     = (_Float16*)d_ws;        // 33.5 MB f16 weights

    // Pass 1: 32768 rows, 8 rows (waves) per block
    const int rows = B_DIM * L_DIM;
    softalign_softmax_kernel<<<rows / 8, 256, 0, stream>>>(attn, mask, Wf16);

    // Pass 2: grid (H/TN, L/TM, B)
    dim3 grid(H_DIM / TN, L_DIM / TM, B_DIM);
    softalign_gemm_kernel<<<grid, 256, 0, stream>>>(Wf16, sentence, out);
}
